// HeteroGAT_59708635349461
// MI455X (gfx1250) — matrix operand order repaired and checked by reading, not compiled, not run on hardware
//
#include <hip/hip_runtime.h>
#include <hip/hip_bf16.h>
#include <stdint.h>

typedef __attribute__((ext_vector_type(16))) _Float16 v16h;
typedef __attribute__((ext_vector_type(8)))  _Float16 v8h;
typedef __attribute__((ext_vector_type(8)))  float    v8f;
typedef int v4i __attribute__((vector_size(4 * sizeof(int))));

#define C_DIM 256
#define NEG_SLOPE 0.2f
#define EPS_LN 1e-5f

#if defined(__has_builtin)
#if __has_builtin(__builtin_amdgcn_global_load_async_to_lds_b128)
#define USE_ASYNC_LDS 1
#endif
#endif

__device__ __forceinline__ void wait_async_lds() {
#if defined(USE_ASYNC_LDS)
#if __has_builtin(__builtin_amdgcn_s_wait_asynccnt)
    __builtin_amdgcn_s_wait_asynccnt(0);
#else
    asm volatile("s_wait_asynccnt 0x0" ::: "memory");
#endif
#endif
}

// ---- order-preserving float<->uint encoding for atomicMax-based segment max
__device__ __forceinline__ unsigned enc_f32(float f) {
    unsigned u = __float_as_uint(f);
    return (u & 0x80000000u) ? ~u : (u | 0x80000000u);
}
__device__ __forceinline__ float dec_f32(unsigned e) {
    unsigned u = (e & 0x80000000u) ? (e & 0x7FFFFFFFu) : ~e;
    return __uint_as_float(u);
}

// ---- transpose W_src (f32 [mat][k][n]) -> WT (f16 [mat][n][k]) for WMMA B-side
__global__ __launch_bounds__(256) void k_wt(const float* __restrict__ W,
                                            _Float16* __restrict__ WT) {
    int b   = blockIdx.x;          // mat*256 + n
    int mat = b >> 8;
    int n   = b & 255;
    int k   = threadIdx.x;
    WT[((size_t)mat * C_DIM + n) * C_DIM + k] =
        (_Float16)W[((size_t)mat * C_DIM + k) * C_DIM + n];
}

// ---- GEMM + fused row-dot:
//   Out[M,256] = A[M,256](f32) x W  (BT = f16 pre-transposed [n][k])
//   aS[r] += dot(Out[r,:], asvec)   (fused in epilogue, aS must be zeroed)
// block = 256 threads = 8 waves; block tile 64 rows x 256 cols;
// wave (wm,wn): 32 rows x 64 cols = 8 WMMA tiles; K-loop 8 steps of 32.
__global__ __launch_bounds__(256) void k_gemm_f16(const float* __restrict__ A,
                                                  const _Float16* __restrict__ BT,
                                                  const float* __restrict__ asvec,
                                                  float* __restrict__ Out,
                                                  float* __restrict__ aS, int M) {
    __shared__ __align__(32) _Float16 As[64][32];
    __shared__ __align__(32) _Float16 Bs[256][32];

    const int tid  = threadIdx.x;
    const int lane = tid & 31;
    const int wave = tid >> 5;
    const int wm   = wave >> 2;   // 0..1 (32-row strip)
    const int wn   = wave & 3;    // 0..3 (64-col strip)
    const int l    = lane & 15;
    const int h    = lane >> 4;   // lane half
    const int rowBase = blockIdx.x * 64;

    v8f acc[2][4] = {};

    const int ar = tid >> 2;        // staged A row 0..63
    const int ac = (tid & 3) * 8;   // staged A col group (8 halfs)

    for (int kk = 0; kk < C_DIM; kk += 32) {
        // stage A tile (f32 -> f16), zero-pad out-of-range rows
        float4 a0 = make_float4(0.f, 0.f, 0.f, 0.f);
        float4 a1 = make_float4(0.f, 0.f, 0.f, 0.f);
        const int gr = rowBase + ar;
        if (gr < M) {
            const float* ap = A + (size_t)gr * C_DIM + kk + ac;
            a0 = *(const float4*)ap;
            a1 = *(const float4*)(ap + 4);
        }
        v8h hv;
        hv[0] = (_Float16)a0.x; hv[1] = (_Float16)a0.y;
        hv[2] = (_Float16)a0.z; hv[3] = (_Float16)a0.w;
        hv[4] = (_Float16)a1.x; hv[5] = (_Float16)a1.y;
        hv[6] = (_Float16)a1.z; hv[7] = (_Float16)a1.w;
        *(v8h*)&As[ar][ac] = hv;

        // stage B tile: thread tid copies 32 halfs (64B) of pre-transposed col n=tid
#if defined(USE_ASYNC_LDS)
        {
            const _Float16* src = BT + (size_t)tid * C_DIM + kk;
            _Float16* dst = &Bs[tid][0];
#pragma unroll
            for (int q = 0; q < 4; ++q) {
                __builtin_amdgcn_global_load_async_to_lds_b128(
                    (__attribute__((address_space(1))) v4i*)(src + q * 8),
                    (__attribute__((address_space(3))) v4i*)(dst + q * 8),
                    0, 0);
            }
        }
        wait_async_lds();
#else
        {
            const uint4* src = (const uint4*)(BT + (size_t)tid * C_DIM + kk);
            uint4* dst = (uint4*)&Bs[tid][0];
            dst[0] = src[0]; dst[1] = src[1]; dst[2] = src[2]; dst[3] = src[3];
        }
#endif
        __syncthreads();

        // A fragments (ISA 16-bit A 16x32 layout): lane l -> K 0..7 & 16..23,
        //                                          lane l+16 -> K 8..15 & 24..31
        v16h afrag[2];
#pragma unroll
        for (int rt = 0; rt < 2; ++rt) {
            const int arow = wm * 32 + rt * 16 + l;
            v8h alo = *(const v8h*)&As[arow][8 * h];
            v8h ahi = *(const v8h*)&As[arow][16 + 8 * h];
            afrag[rt] = __builtin_shufflevector(alo, ahi, 0, 1, 2, 3, 4, 5, 6, 7,
                                                8, 9, 10, 11, 12, 13, 14, 15);
        }
#pragma unroll
        for (int i = 0; i < 4; ++i) {
            // B fragment: lane l col N, K 0..7 & 16..23 (h=0) / 8..15 & 24..31 (h=1)
            const int bcol = wn * 64 + i * 16 + l;
            v8h blo = *(const v8h*)&Bs[bcol][8 * h];
            v8h bhi = *(const v8h*)&Bs[bcol][16 + 8 * h];
            v16h bfrag = __builtin_shufflevector(blo, bhi, 0, 1, 2, 3, 4, 5, 6, 7,
                                                 8, 9, 10, 11, 12, 13, 14, 15);
#pragma unroll
            for (int rt = 0; rt < 2; ++rt) {
                acc[rt][i] = __builtin_amdgcn_wmma_f32_16x16x32_f16(
                    false, afrag[rt], false, bfrag, (short)0, acc[rt][i], false, false);
            }
        }
        __syncthreads();
    }

    // per-lane a_s values for this wave's 4 column tiles
    float asv[4];
#pragma unroll
    for (int i = 0; i < 4; ++i) asv[i] = asvec[wn * 64 + i * 16 + l];

    // epilogue: D layout — VGPR j, lane half h: row = 8*h + j, col = lane&15
#pragma unroll
    for (int rt = 0; rt < 2; ++rt) {
        const int orow0 = rowBase + wm * 32 + rt * 16 + 8 * h;
        float p[8] = {0.f, 0.f, 0.f, 0.f, 0.f, 0.f, 0.f, 0.f};
#pragma unroll
        for (int i = 0; i < 4; ++i) {
            const int col = wn * 64 + i * 16 + l;
#pragma unroll
            for (int j = 0; j < 8; ++j) {
                const int r = orow0 + j;
                const float d = acc[rt][i][j];
                if (r < M) Out[(size_t)r * C_DIM + col] = d;
                p[j] += d * asv[i];
            }
        }
        // reduce partial dots across the 16 lanes of this half (same rows)
#pragma unroll
        for (int j = 0; j < 8; ++j) {
            p[j] += __shfl_xor(p[j], 1, 32);
            p[j] += __shfl_xor(p[j], 2, 32);
            p[j] += __shfl_xor(p[j], 4, 32);
            p[j] += __shfl_xor(p[j], 8, 32);
        }
        if (l == 0) {
#pragma unroll
            for (int j = 0; j < 8; ++j) {
                const int r = orow0 + j;
                if (r < M) atomicAdd(aS + r, p[j]);
            }
        }
    }
}

// ---- out[r] = dot(X[r,:], vec)   (wave per row)
__global__ __launch_bounds__(256) void k_rowdot(const float* __restrict__ X,
                                                const float* __restrict__ vec,
                                                float* __restrict__ out, int R) {
    int wave = threadIdx.x >> 5, lane = threadIdx.x & 31;
    int row = blockIdx.x * 8 + wave;
    if (row >= R) return;
    const float* x = X + (size_t)row * C_DIM;
    float s = 0.f;
#pragma unroll
    for (int i = 0; i < 8; ++i) { int c = lane + i * 32; s += x[c] * vec[c]; }
#pragma unroll
    for (int o = 16; o > 0; o >>= 1) s += __shfl_xor(s, o, 32);
    if (lane == 0) out[row] = s;
}

__global__ void k_zero_f32(float* __restrict__ p, size_t n) {
    size_t i = (size_t)blockIdx.x * blockDim.x + threadIdx.x;
    if (i < n) p[i] = 0.f;
}
__global__ void k_init_reduce(unsigned* __restrict__ menc,
                              float* __restrict__ den, int n) {
    int i = blockIdx.x * blockDim.x + threadIdx.x;
    if (i < n) { menc[i] = 0x007FFFFFu; /* enc(-inf) */ den[i] = 0.f; }
}

__global__ void k_edge_logits(const int* __restrict__ ei, int E,
                              const float* __restrict__ aS,
                              const float* __restrict__ aD,
                              float* __restrict__ ev,
                              unsigned* __restrict__ menc) {
    int e = blockIdx.x * blockDim.x + threadIdx.x;
    if (e >= E) return;
    int s = ei[e], d = ei[E + e];
    float v = aS[s] + aD[d];
    v = v > 0.f ? v : NEG_SLOPE * v;
    ev[e] = v;
    atomicMax(menc + d, enc_f32(v));
}

__global__ void k_edge_den(const int* __restrict__ ei, int E,
                           const float* __restrict__ ev,
                           const unsigned* __restrict__ menc,
                           float* __restrict__ den) {
    int e = blockIdx.x * blockDim.x + threadIdx.x;
    if (e >= E) return;
    int d = ei[E + e];
    atomicAdd(den + d, __expf(ev[e] - dec_f32(menc[d])));
}

// one block (256 threads) per edge: out[dst,:] += coef * H[src,:]
__global__ __launch_bounds__(256) void k_edge_scatter(const int* __restrict__ ei, int E,
                                                      const float* __restrict__ ev,
                                                      const unsigned* __restrict__ menc,
                                                      const float* __restrict__ den,
                                                      const float* __restrict__ H,
                                                      float* __restrict__ Out) {
    int e = blockIdx.x;
    int s = ei[e], d = ei[E + e];
    float coef = __expf(ev[e] - dec_f32(menc[d])) / den[d];
    int c = threadIdx.x;
    atomicAdd(Out + (size_t)d * C_DIM + c, coef * H[(size_t)s * C_DIM + c]);
}

// fused (x + b1 + w2*b2) -> LayerNorm -> *g+b -> ReLU, in place, wave per row
__global__ __launch_bounds__(256) void k_ln_relu(float* __restrict__ X,
                                                 const float* __restrict__ b1,
                                                 const float* __restrict__ b2, float w2,
                                                 const float* __restrict__ g,
                                                 const float* __restrict__ bv, int R) {
    int wave = threadIdx.x >> 5, lane = threadIdx.x & 31;
    int row = blockIdx.x * 8 + wave;
    if (row >= R) return;
    float* x = X + (size_t)row * C_DIM;
    float v[8], s = 0.f, s2 = 0.f;
#pragma unroll
    for (int i = 0; i < 8; ++i) {
        int c = lane + i * 32;
        float t = x[c] + b1[c] + w2 * b2[c];
        v[i] = t; s += t; s2 += t * t;
    }
#pragma unroll
    for (int o = 16; o > 0; o >>= 1) { s += __shfl_xor(s, o, 32); s2 += __shfl_xor(s2, o, 32); }
    float mu  = s * (1.f / C_DIM);
    float var = s2 * (1.f / C_DIM) - mu * mu;
    float rs  = rsqrtf(var + EPS_LN);
#pragma unroll
    for (int i = 0; i < 8; ++i) {
        int c = lane + i * 32;
        float y = (v[i] - mu) * rs * g[c] + bv[c];
        x[c] = fmaxf(y, 0.f);
    }
}

extern "C" void kernel_launch(void* const* d_in, const int* in_sizes, int n_in,
                              void* d_out, int out_size, void* d_ws, size_t ws_size,
                              hipStream_t stream) {
    const float* x_user  = (const float*)d_in[0];
    const float* x_item  = (const float*)d_in[1];
    const float* W_src   = (const float*)d_in[2];
    const float* W_dst   = (const float*)d_in[3];
    const float* att_src = (const float*)d_in[4];
    const float* att_dst = (const float*)d_in[5];
    const float* bias    = (const float*)d_in[6];
    const float* ln_g    = (const float*)d_in[7];
    const float* ln_b    = (const float*)d_in[8];
    const int* ei[3] = { (const int*)d_in[9], (const int*)d_in[10], (const int*)d_in[11] };
    int Ecnt[3] = { in_sizes[9] / 2, in_sizes[10] / 2, in_sizes[11] / 2 };

    const int Nu = in_sizes[0] / C_DIM;
    const int Ni = in_sizes[1] / C_DIM;
    const int Nmax = Nu > Ni ? Nu : Ni;
    int Emax = Ecnt[0];
    if (Ecnt[1] > Emax) Emax = Ecnt[1];
    if (Ecnt[2] > Emax) Emax = Ecnt[2];

    const size_t NCu = (size_t)Nu * C_DIM, NCi = (size_t)Ni * C_DIM;
    const size_t NCmax = (size_t)Nmax * C_DIM;

    // workspace carve
    float* H      = (float*)d_ws;
    float* U1     = H + NCmax;
    float* I1     = U1 + NCu;
    float* aS     = I1 + NCi;
    float* aD     = aS + Nmax;
    unsigned* menc = (unsigned*)(aD + Nmax);
    float* den    = (float*)(menc + Nmax);
    float* ev     = den + Nmax;
    float* wdv    = ev + Emax;
    _Float16* WT  = (_Float16*)(wdv + C_DIM);   // 6 * 256 * 256 halfs

    // pre-transpose the 6 source weight matrices to f16 (B side of WMMA GEMMs)
    k_wt<<<6 * 256, 256, 0, stream>>>(W_src, WT);

    float* out_u = (float*)d_out;
    float* out_i = (float*)d_out + NCu;

    for (int layer = 0; layer < 2; ++layer) {
        const float* xu = (layer == 0) ? x_user : U1;
        const float* xi = (layer == 0) ? x_item : I1;
        float* ou = (layer == 0) ? U1 : out_u;
        float* oi = (layer == 0) ? I1 : out_i;

        k_zero_f32<<<(unsigned)((NCu + 255) / 256), 256, 0, stream>>>(ou, NCu);
        k_zero_f32<<<(unsigned)((NCi + 255) / 256), 256, 0, stream>>>(oi, NCi);

        for (int t = 0; t < 3; ++t) {
            const float* xsrc; const float* xdst;
            int Ms, Md; float* outp;
            if (t == 0)      { xsrc = xu; Ms = Nu; xdst = xi; Md = Ni; outp = oi; }
            else if (t == 1) { xsrc = xi; Ms = Ni; xdst = xu; Md = Nu; outp = ou; }
            else             { xsrc = xu; Ms = Nu; xdst = xu; Md = Nu; outp = ou; }
            const int mi = layer * 3 + t;
            const int E = Ecnt[t];

            // hs = x_src @ W_src[l,t] with fused alpha_s = hs @ a_s
            k_zero_f32<<<(Ms + 255) / 256, 256, 0, stream>>>(aS, (size_t)Ms);
            k_gemm_f16<<<(Ms + 63) / 64, 256, 0, stream>>>(
                xsrc, WT + (size_t)mi * C_DIM * C_DIM,
                att_src + (size_t)mi * C_DIM, H, aS, Ms);
            // wdvec = W_dst[l,t] @ a_d  (row-dot of W_dst rows)
            k_rowdot<<<(C_DIM + 7) / 8, 256, 0, stream>>>(
                W_dst + (size_t)mi * C_DIM * C_DIM, att_dst + (size_t)mi * C_DIM, wdv, C_DIM);
            // alpha_d = x_dst @ wdvec   (skips the full hd GEMM entirely)
            k_rowdot<<<(Md + 7) / 8, 256, 0, stream>>>(xdst, wdv, aD, Md);

            k_init_reduce<<<(Md + 255) / 256, 256, 0, stream>>>(menc, den, Md);
            k_edge_logits<<<(E + 255) / 256, 256, 0, stream>>>(ei[t], E, aS, aD, ev, menc);
            k_edge_den<<<(E + 255) / 256, 256, 0, stream>>>(ei[t], E, ev, menc, den);
            k_edge_scatter<<<E, 256, 0, stream>>>(ei[t], E, ev, menc, den, H, outp);
        }

        // user: conv(i2u)+bias1 + conv(u2u)+bias2 -> LN(g0,b0) -> ReLU
        k_ln_relu<<<(Nu + 7) / 8, 256, 0, stream>>>(
            ou, bias + (size_t)(layer * 3 + 1) * C_DIM, bias + (size_t)(layer * 3 + 2) * C_DIM,
            1.f, ln_g + (size_t)(layer * 2 + 0) * C_DIM, ln_b + (size_t)(layer * 2 + 0) * C_DIM, Nu);
        // item: conv(u2i)+bias0 -> LN(g1,b1) -> ReLU
        k_ln_relu<<<(Ni + 7) / 8, 256, 0, stream>>>(
            oi, bias + (size_t)(layer * 3 + 0) * C_DIM, bias + (size_t)(layer * 3 + 0) * C_DIM,
            0.f, ln_g + (size_t)(layer * 2 + 1) * C_DIM, ln_b + (size_t)(layer * 2 + 1) * C_DIM, Ni);
    }
    (void)n_in; (void)out_size; (void)ws_size;
}
